// AttentionModule_70136815943908
// MI455X (gfx1250) — matrix operand order repaired and checked by reading, not compiled, not run on hardware
//
#include <hip/hip_runtime.h>
#include <cstdint>

// ---------------------------------------------------------------- constants
static constexpr int BB = 32;            // batch
static constexpr int CC = 512;           // channels
static constexpr int NN = 1024;          // H*W
static constexpr int DD = 64;            // C/8
static constexpr int CN = CC * NN;       // per-batch elems
static constexpr int WROWS = DD + CC;    // 576 concatenated weight rows

// ---------------------------------------------------------------- WMMA types
typedef __attribute__((ext_vector_type(16))) __bf16 v16bf;
typedef __attribute__((ext_vector_type(8)))  float  v8f;

union BFrag { v16bf v; uint32_t u[8]; };

__device__ __forceinline__ v8f wmma_bf16(const BFrag& a, const BFrag& b, v8f c) {
    // (neg_a, A, neg_b, B, c_mod, C, reuse_a, reuse_b)
    return __builtin_amdgcn_wmma_f32_16x16x32_bf16(false, a.v, false, b.v,
                                                   (short)0, c, false, false);
}

// ---------------------------------------------------------------- bf16 cvt
// Let the backend pick the native f32->bf16 convert on gfx1250.
__device__ __forceinline__ uint16_t bf16_1(float f) {
    __bf16 h = (__bf16)f;
    return __builtin_bit_cast(uint16_t, h);
}
__device__ __forceinline__ uint32_t pack_bf16x2(float lo, float hi) {
    return (uint32_t)bf16_1(lo) | ((uint32_t)bf16_1(hi) << 16);
}

// ------------------------------------------------- fragment loaders (wave32)
// A frag in this file's contiguous k order (B matches): lanes 0-15 M=lane, K 0..15;
// lanes 16-31 same M, K 16..31.  Expressed as two 16B loads.
__device__ __forceinline__ void load_a_frag(BFrag& f, const uint16_t* rowbase, int hi) {
    const uint4* p = (const uint4*)rowbase;
    *(uint4*)&f.u[0] = p[2 * hi];        // K {0..7}  (+16 if hi)
    *(uint4*)&f.u[4] = p[2 * hi + 1];    // K {8..15} (+16 if hi)
}
// B-matrix 32x16 bf16: lane = N (+16 for K-half), 16 contiguous bf16 per lane.
// base must already include the +16*hi K offset.
__device__ __forceinline__ void load_b_frag(BFrag& f, const uint16_t* base) {
    const uint4* p = (const uint4*)base;
    *(uint4*)&f.u[0] = p[0];
    *(uint4*)&f.u[4] = p[1];
}

// ================================================================ kernel 0:
// Weight prep (runs once, tiny): Wcat bf16 [576,512] = [w1; w2], bcat f32[576].
__global__ __launch_bounds__(256) void k_wprep(const float* __restrict__ w1,
                                               const float* __restrict__ b1,
                                               const float* __restrict__ w2,
                                               const float* __restrict__ b2,
                                               uint16_t* __restrict__ Wcat,
                                               float* __restrict__ bcat) {
    int idx = blockIdx.x * 256 + threadIdx.x;
    if (idx < WROWS * CC) {
        int row = idx / CC, col = idx % CC;
        float val = (row < DD) ? w1[row * CC + col] : w2[(row - DD) * CC + col];
        Wcat[idx] = bf16_1(val);
    }
    if (idx < WROWS) bcat[idx] = (idx < DD) ? b1[idx] : b2[idx - DD];
}

// ================================================================ kernel 1:
// x [B,C,N] f32  ->  xT [B,N,C] bf16  (tiled transpose + convert, packed
// dword stores).  Block (16,16), tile 32 channels x 16 columns.
__global__ __launch_bounds__(256) void k_transpose(const float* __restrict__ x,
                                                   uint16_t* __restrict__ xT) {
    __shared__ float tile[32][17];
    int b = blockIdx.z, c0 = blockIdx.y * 32, n0 = blockIdx.x * 16;
    int tx = threadIdx.x, ty = threadIdx.y;
    tile[ty][tx]      = x[b * CN + (c0 + ty) * NN + n0 + tx];
    tile[ty + 16][tx] = x[b * CN + (c0 + ty + 16) * NN + n0 + tx];
    __syncthreads();
    uint32_t pk = pack_bf16x2(tile[2 * tx][ty], tile[2 * tx + 1][ty]);
    *(uint32_t*)&xT[b * NN * CC + (n0 + ty) * CC + c0 + 2 * tx] = pk;
}

// ================================================================ kernel 2:
// dual GEMM from pre-converted bf16 weights:
//   qkT[b,n,d] = (w1 . x + b1)^T   (bf16, [B,N,D])
//   v  [b,c,n] =  w2 . x + b2      (bf16, [B,C,N])
// grid (N/16, B), block 256 (8 waves). Each wave stages the 16 B-frags of its
// n-tile of xT once, then sweeps row tiles rt = wave, wave+8, ... < 36 (global row = rt*16+m).
__global__ __launch_bounds__(256) void k_gemm(const uint16_t* __restrict__ xT,
                                              const uint16_t* __restrict__ Wcat,
                                              const float* __restrict__ bcat,
                                              uint16_t* __restrict__ qkT,
                                              uint16_t* __restrict__ v) {
    int b = blockIdx.y, n0 = blockIdx.x * 16;
    int tid = threadIdx.x, wave = tid >> 5, lane = tid & 31;
    int hi = lane >> 4, l16 = lane & 15;

    const uint16_t* xb = xT + b * NN * CC;

    // Stage B fragments for all K=512 (16 steps of 32) of this n-tile.
    BFrag bfr[16];
#pragma unroll
    for (int ks = 0; ks < 16; ++ks)
        load_b_frag(bfr[ks], xb + (n0 + l16) * CC + ks * 32 + 16 * hi);

    for (int rt = wave; rt < 36; rt += 8) {
        const uint16_t* Wrow = Wcat + (rt * 16 + l16) * CC;

        v8f acc = {0.f, 0.f, 0.f, 0.f, 0.f, 0.f, 0.f, 0.f};
#pragma unroll
        for (int ks = 0; ks < 16; ++ks) {
            BFrag af;
            load_a_frag(af, Wrow + ks * 32, hi);
            acc = wmma_bf16(af, bfr[ks], acc);
        }
        // C/D layout: VGPR j, lanes 0-15 -> M=j, N=lane; lanes 16-31 -> M=j+8.
        if (rt < 4) {   // w1 rows -> qkT (transposed store)
#pragma unroll
            for (int j = 0; j < 8; ++j) {
                int row = rt * 16 + j + 8 * hi;         // d index
                int col = n0 + l16;                     // n index
                qkT[b * NN * DD + col * DD + row] = bf16_1(acc[j] + bcat[row]);
            }
        } else {        // w2 rows -> v [C,N]
#pragma unroll
            for (int j = 0; j < 8; ++j) {
                int grow = rt * 16 + j + 8 * hi;        // 64 + c index
                int crow = grow - DD;
                v[b * CC * NN + crow * NN + n0 + l16] = bf16_1(acc[j] + bcat[grow]);
            }
        }
    }
}

// ================================================================ kernel 3:
// Fused attention: per (b, 16-row n-tile):
//   S = qkT[rows] . qk  (16x1024, in WMMA accs, 8 tiles per wave)
//   softmax (register shuffles + LDS cross-wave reduce), P -> LDS bf16
//   out = P . v^T  (16x512), scale by 1/rowsum, residual add with the
//   reference's flat-view quirk: d_out[b, n*512+c] = x[same] + out[n][c].
__global__ __launch_bounds__(256) void k_attn(const float* __restrict__ x,
                                              const uint16_t* __restrict__ qkT,
                                              const uint16_t* __restrict__ v,
                                              float* __restrict__ out) {
    __shared__ uint16_t Pbuf[16 * 1024];   // 32 KB unnormalized exp scores (bf16)
    __shared__ float red[8][16];
    __shared__ float rowmax[16];
    __shared__ float rowinv[16];

    int b = blockIdx.y, n0 = blockIdx.x * 16;
    int tid = threadIdx.x, wave = tid >> 5, lane = tid & 31;
    int hi = lane >> 4, l16 = lane & 15;

    const uint16_t* qb = qkT + b * NN * DD;

    // A frags: 16 query rows x K=64 (two K=32 fragments), shared by all tiles.
    BFrag a0, a1;
    load_a_frag(a0, qb + (n0 + l16) * DD + 0,  hi);
    load_a_frag(a1, qb + (n0 + l16) * DD + 32, hi);

    // ---- phase 1: scores. Wave w owns m-tiles w*8 .. w*8+7 (cols m0..m0+15).
    v8f s[8];
#pragma unroll
    for (int t = 0; t < 8; ++t) {
        int m0 = (wave * 8 + t) * 16;
        BFrag kb0, kb1;
        load_b_frag(kb0, qb + (m0 + l16) * DD + 0  + 16 * hi);
        load_b_frag(kb1, qb + (m0 + l16) * DD + 32 + 16 * hi);
        v8f z = {0.f, 0.f, 0.f, 0.f, 0.f, 0.f, 0.f, 0.f};
        z = wmma_bf16(a0, kb0, z);
        s[t] = wmma_bf16(a1, kb1, z);
    }

    // ---- row max: elementwise over tiles, xor-shuffle within 16-lane halves,
    // then cross-wave reduce through LDS.
    float m8[8];
#pragma unroll
    for (int j = 0; j < 8; ++j) {
        float m = s[0][j];
#pragma unroll
        for (int t = 1; t < 8; ++t) m = fmaxf(m, s[t][j]);
#pragma unroll
        for (int off = 8; off >= 1; off >>= 1)
            m = fmaxf(m, __shfl_xor(m, off, 32));
        m8[j] = m;
    }
    if (l16 == 0) {
#pragma unroll
        for (int j = 0; j < 8; ++j) red[wave][j + 8 * hi] = m8[j];
    }
    __syncthreads();
    if (tid < 16) {
        float m = red[0][tid];
#pragma unroll
        for (int w = 1; w < 8; ++w) m = fmaxf(m, red[w][tid]);
        rowmax[tid] = m;
    }
    __syncthreads();

    // ---- exp, partial row sums, P -> LDS (bf16, unnormalized).
    float sum8[8] = {0.f, 0.f, 0.f, 0.f, 0.f, 0.f, 0.f, 0.f};
#pragma unroll
    for (int t = 0; t < 8; ++t) {
        int col = (wave * 8 + t) * 16 + l16;
#pragma unroll
        for (int j = 0; j < 8; ++j) {
            int row = j + 8 * hi;
            float e = __expf(s[t][j] - rowmax[row]);
            sum8[j] += e;
            Pbuf[row * 1024 + col] = bf16_1(e);
        }
    }
#pragma unroll
    for (int j = 0; j < 8; ++j) {
        float sm = sum8[j];
#pragma unroll
        for (int off = 8; off >= 1; off >>= 1) sm += __shfl_xor(sm, off, 32);
        sum8[j] = sm;
    }
    if (l16 == 0) {
#pragma unroll
        for (int j = 0; j < 8; ++j) red[wave][j + 8 * hi] = sum8[j];
    }
    __syncthreads();
    if (tid < 16) {
        float sm = 0.f;
#pragma unroll
        for (int w = 0; w < 8; ++w) sm += red[w][tid];
        rowinv[tid] = 1.0f / sm;
    }
    __syncthreads();

    // ---- phase 2: out = P . v^T.  Wave owns 4 c-tiles; K = 1024 in 32 steps.
    const uint16_t* vb = v + b * CC * NN;
    v8f o[4];
#pragma unroll
    for (int ct = 0; ct < 4; ++ct) o[ct] = (v8f){0.f,0.f,0.f,0.f,0.f,0.f,0.f,0.f};

    for (int ks = 0; ks < 32; ++ks) {
        int m0 = ks * 32;
        BFrag ap;
        load_a_frag(ap, &Pbuf[l16 * 1024 + m0], hi);        // LDS reads (ds)
#pragma unroll
        for (int ct = 0; ct < 4; ++ct) {
            int c0 = (wave * 4 + ct) * 16;
            BFrag bv;
            load_b_frag(bv, vb + (c0 + l16) * NN + m0 + 16 * hi);
            o[ct] = wmma_bf16(ap, bv, o[ct]);
        }
    }

    // ---- store with softmax scale + residual, using the flat-view quirk.
#pragma unroll
    for (int ct = 0; ct < 4; ++ct) {
        int c = (wave * 4 + ct) * 16 + l16;
#pragma unroll
        for (int j = 0; j < 8; ++j) {
            int row = j + 8 * hi;
            int idx = b * CN + (n0 + row) * CC + c;
            out[idx] = x[idx] + o[ct][j] * rowinv[row];
        }
    }
}

// ================================================================ launcher
extern "C" void kernel_launch(void* const* d_in, const int* in_sizes, int n_in,
                              void* d_out, int out_size, void* d_ws, size_t ws_size,
                              hipStream_t stream) {
    const float* x  = (const float*)d_in[0];   // [B,C,H,W]
    const float* w1 = (const float*)d_in[1];   // [D,C]
    const float* b1 = (const float*)d_in[2];   // [D]
    const float* w2 = (const float*)d_in[3];   // [C,C]
    const float* b2 = (const float*)d_in[4];   // [C]
    float* out = (float*)d_out;

    // Workspace layout: xT bf16 32MB | qkT bf16 4MB | v bf16 32MB |
    //                   Wcat bf16 576KB | bcat f32 2.3KB
    uint16_t* xT   = (uint16_t*)d_ws;
    uint16_t* qkT  = xT  + (size_t)BB * NN * CC;
    uint16_t* v    = qkT + (size_t)BB * NN * DD;
    uint16_t* Wcat = v   + (size_t)BB * CC * NN;
    float*    bcat = (float*)(Wcat + (size_t)WROWS * CC);

    k_wprep<<<(WROWS * CC + 255) / 256, 256, 0, stream>>>(w1, b1, w2, b2, Wcat, bcat);
    k_transpose<<<dim3(NN / 16, CC / 32, BB), dim3(16, 16), 0, stream>>>(x, xT);
    k_gemm<<<dim3(NN / 16, BB), 256, 0, stream>>>(xT, Wcat, bcat, qkT, v);
    k_attn<<<dim3(NN / 16, BB), 256, 0, stream>>>(x, qkT, v, out);
}
